// selfAttentionV1_11647951306944
// MI455X (gfx1250) — matrix operand order repaired and checked by reading, not compiled, not run on hardware
//
#include <hip/hip_runtime.h>

// ---------------------------------------------------------------------------
// Self-attention for MI455X (gfx1250): bf16 WMMA, f32 accumulate.
// All GEMMs run in NT form (B pre-transposed), so global->LDS staging is pure
// b128 copies with loads issued a full stage ahead (double-buffered LDS).
//   SEQ = 4096, D_IN = D_OUT = 1024
// ---------------------------------------------------------------------------

#define SEQ  4096
#define DIN  1024
#define DOUT 1024

typedef __bf16          v16bf __attribute__((ext_vector_type(16)));
typedef __bf16          v8bf  __attribute__((ext_vector_type(8)));
typedef float           v8f   __attribute__((ext_vector_type(8)));
typedef unsigned short  v8us  __attribute__((ext_vector_type(8)));
typedef unsigned short  v4us  __attribute__((ext_vector_type(4)));

__device__ __forceinline__ unsigned short f2bf(float f) {
  unsigned int u = __float_as_uint(f);
  u += 0x7FFFu + ((u >> 16) & 1u);
  return (unsigned short)(u >> 16);
}

// ---------------------------------------------------------------------------
// fp32 -> bf16 bulk convert, 8 elements / thread
// ---------------------------------------------------------------------------
__global__ void cvt_f32_bf16(const float* __restrict__ src,
                             unsigned short* __restrict__ dst, int n8) {
  int i = blockIdx.x * blockDim.x + threadIdx.x;
  if (i >= n8) return;
  const float4* s = (const float4*)src + 2 * (size_t)i;
  float4 a = s[0], b = s[1];
  v8us o;
  o[0] = f2bf(a.x); o[1] = f2bf(a.y); o[2] = f2bf(a.z); o[3] = f2bf(a.w);
  o[4] = f2bf(b.x); o[5] = f2bf(b.y); o[6] = f2bf(b.z); o[7] = f2bf(b.w);
  *((v8us*)dst + i) = o;
}

// ---------------------------------------------------------------------------
// fp32 [R][C] -> bf16 transposed [C][R].  64x64 LDS tile, coalesced both sides.
// ---------------------------------------------------------------------------
__global__ __launch_bounds__(256)
void transpose_cvt_f32_bf16(const float* __restrict__ src,
                            unsigned short* __restrict__ dst, int R, int C) {
  __shared__ __align__(16) unsigned short T[64][72];
  const int tid = threadIdx.x;
  const int rt = blockIdx.y * 64, ct = blockIdx.x * 64;
#pragma unroll
  for (int p = 0; p < 4; ++p) {
    int g = tid + p * 256;
    int r = g >> 4, c4 = (g & 15) * 4;
    float4 v = *(const float4*)(src + (size_t)(rt + r) * C + ct + c4);
    v4us o;
    o[0] = f2bf(v.x); o[1] = f2bf(v.y); o[2] = f2bf(v.z); o[3] = f2bf(v.w);
    *(v4us*)&T[r][c4] = o;
  }
  __syncthreads();
#pragma unroll
  for (int p = 0; p < 2; ++p) {
    int g = tid + p * 256;
    int rr = g >> 3, cc0 = (g & 7) * 8;   // rr: col of src tile, cc0: row group
    v8us o;
#pragma unroll
    for (int j = 0; j < 8; ++j) o[j] = T[cc0 + j][rr];
    *(v8us*)(dst + (size_t)(ct + rr) * R + rt + cc0) = o;
  }
}

// ---------------------------------------------------------------------------
// bf16 [R][C] -> bf16 transposed [C][R].  64x64 LDS tile.
// ---------------------------------------------------------------------------
__global__ __launch_bounds__(256)
void transpose_bf16(const unsigned short* __restrict__ src,
                    unsigned short* __restrict__ dst, int R, int C) {
  __shared__ __align__(16) unsigned short T[64][72];
  const int tid = threadIdx.x;
  const int rt = blockIdx.y * 64, ct = blockIdx.x * 64;
#pragma unroll
  for (int p = 0; p < 2; ++p) {
    int g = tid + p * 256;
    int r = g >> 3, c8 = (g & 7) * 8;
    *(v8us*)&T[r][c8] = *(const v8us*)(src + (size_t)(rt + r) * C + ct + c8);
  }
  __syncthreads();
#pragma unroll
  for (int p = 0; p < 2; ++p) {
    int g = tid + p * 256;
    int rr = g >> 3, cc0 = (g & 7) * 8;
    v8us o;
#pragma unroll
    for (int j = 0; j < 8; ++j) o[j] = T[cc0 + j][rr];
    *(v8us*)(dst + (size_t)(ct + rr) * R + rt + cc0) = o;
  }
}

// ---------------------------------------------------------------------------
// bf16 WMMA GEMM, NT form: C[M,N] = scale * A[M,K] x B[N,K]^T
// (A row-major lda, B row-major ldb over K; both staged with b128 copies)
// 256 threads (8 wave32), 128x128 block tile, Kc=64/stage, double-buffered
// LDS, register-staged copies with loads issued one full stage ahead.
// Per wave: 32x64 sub-tile, 2 k-steps x 8 v_wmma = 16 WMMAs per stage.
// ---------------------------------------------------------------------------
template <bool OUT_BF16>
__global__ __launch_bounds__(256)
void gemm_nt_bf16_wmma(const unsigned short* __restrict__ A, int lda,
                       const unsigned short* __restrict__ B, int ldb,
                       void* __restrict__ Cout, int ldc, int K, float scale) {
  __shared__ __align__(16) unsigned short As[2][128][72];  // [buf][m][k]+8 pad
  __shared__ __align__(16) unsigned short Bs[2][128][72];  // [buf][n][k]+8 pad

  const int tid  = threadIdx.x;
  const int wave = tid >> 5;
  const int lane = tid & 31;
  const int l15  = lane & 15;
  const int lh   = lane >> 4;          // K-half select (WMMA operand layout)

  const int mblk = blockIdx.y * 128;
  const int nblk = blockIdx.x * 128;
  const int m_off = 32 * (wave & 3);   // 4 waves down
  const int n_off = 64 * (wave >> 2);  // 2 waves across

  // staging: tile 128x64 -> 4 passes of (row = tid/8 + 32p, k = (tid%8)*8)
  const int r0 = tid >> 3;
  const int k0 = (tid & 7) * 8;

  const unsigned short* aptr[4];
  const unsigned short* bptr[4];
#pragma unroll
  for (int p = 0; p < 4; ++p) {
    aptr[p] = A + (size_t)(mblk + r0 + 32 * p) * lda + k0;
    bptr[p] = B + (size_t)(nblk + r0 + 32 * p) * ldb + k0;
  }

  v8f acc[2][4];
#pragma unroll
  for (int i = 0; i < 2; ++i)
#pragma unroll
    for (int j = 0; j < 4; ++j) {
      v8f z = {0.f, 0.f, 0.f, 0.f, 0.f, 0.f, 0.f, 0.f};
      acc[i][j] = z;
    }

  // ---- prologue: first K-chunk into registers (8 loads in flight) ----
  v8us ar[4], br[4];
#pragma unroll
  for (int p = 0; p < 4; ++p) { ar[p] = *(const v8us*)aptr[p]; aptr[p] += 64; }
#pragma unroll
  for (int p = 0; p < 4; ++p) { br[p] = *(const v8us*)bptr[p]; bptr[p] += 64; }

  int buf = 0;
  for (int kc = 0; kc < K; kc += 64) {
    // ---- drain staged registers into LDS[buf] (pure b128) ----
#pragma unroll
    for (int p = 0; p < 4; ++p)
      *(v8us*)&As[buf][r0 + 32 * p][k0] = ar[p];
#pragma unroll
    for (int p = 0; p < 4; ++p)
      *(v8us*)&Bs[buf][r0 + 32 * p][k0] = br[p];
    __syncthreads();

    // ---- issue next chunk's loads; their wait lands at next ds_store ----
    if (kc + 64 < K) {
#pragma unroll
      for (int p = 0; p < 4; ++p) { ar[p] = *(const v8us*)aptr[p]; aptr[p] += 64; }
#pragma unroll
      for (int p = 0; p < 4; ++p) { br[p] = *(const v8us*)bptr[p]; bptr[p] += 64; }
      if (kc + 128 < K) {  // pull the chunk after that toward L2
        __builtin_prefetch(aptr[0], 0, 1);
        __builtin_prefetch(bptr[0], 0, 1);
      }
    }

    // ---- compute: 2 k-steps of 8 WMMAs from LDS[buf] ----
#pragma unroll
    for (int ks = 0; ks < 2; ++ks) {
      const int ko = ks * 32;
      v16bf a[2], b[4];
#pragma unroll
      for (int i = 0; i < 2; ++i) {
        const unsigned short* r = &As[buf][m_off + 16 * i + l15][ko + 8 * lh];
        v8bf lo = *(const v8bf*)r;        // K = ko+8*lh    .. +8
        v8bf hi = *(const v8bf*)(r + 16); // K = ko+16+8*lh .. +8
        a[i] = __builtin_shufflevector(lo, hi, 0, 1, 2, 3, 4, 5, 6, 7, 8, 9,
                                       10, 11, 12, 13, 14, 15);
      }
#pragma unroll
      for (int j = 0; j < 4; ++j) {
        const unsigned short* r = &Bs[buf][n_off + 16 * j + l15][ko + 8 * lh];
        v8bf lo = *(const v8bf*)r;        // K = ko+8*lh    .. +8
        v8bf hi = *(const v8bf*)(r + 16); // K = ko+16+8*lh .. +8
        b[j] = __builtin_shufflevector(lo, hi, 0, 1, 2, 3, 4, 5, 6, 7, 8, 9,
                                       10, 11, 12, 13, 14, 15);
      }
#pragma unroll
      for (int i = 0; i < 2; ++i)
#pragma unroll
        for (int j = 0; j < 4; ++j)
          acc[i][j] = __builtin_amdgcn_wmma_f32_16x16x32_bf16(
              false, a[i], false, b[j], (short)0, acc[i][j], false, false);
    }
    buf ^= 1;
  }

  // ---- store C: lane holds column n, rows m_off+16i+8*lh+e ----
#pragma unroll
  for (int i = 0; i < 2; ++i)
#pragma unroll
    for (int j = 0; j < 4; ++j) {
      int n = nblk + n_off + 16 * j + l15;
      int rbase = mblk + m_off + 16 * i + 8 * lh;
#pragma unroll
      for (int e = 0; e < 8; ++e) {
        float v = acc[i][j][e] * scale;
        if (OUT_BF16)
          ((unsigned short*)Cout)[(size_t)(rbase + e) * ldc + n] = f2bf(v);
        else
          ((float*)Cout)[(size_t)(rbase + e) * ldc + n] = v;
      }
    }
}

// ---------------------------------------------------------------------------
// Row softmax, in place: read 4096 f32 (16/thread, kept in registers),
// reduce max & sum via LDS, write 4096 bf16 over the front of the same row.
// Safe: all global loads complete before the first __syncthreads().
// ---------------------------------------------------------------------------
__global__ __launch_bounds__(256)
void softmax_row_inplace(float* __restrict__ S, int n) {
  __shared__ float red[256];
  const int t = threadIdx.x;
  float* rp = S + (size_t)blockIdx.x * n;

  float v[16];
  const float4* p4 = (const float4*)rp + t * 4;
#pragma unroll
  for (int q = 0; q < 4; ++q) {
    float4 x = p4[q];
    v[4 * q + 0] = x.x; v[4 * q + 1] = x.y;
    v[4 * q + 2] = x.z; v[4 * q + 3] = x.w;
  }
  float m = v[0];
#pragma unroll
  for (int i = 1; i < 16; ++i) m = fmaxf(m, v[i]);
  red[t] = m;
  __syncthreads();
  for (int s = 128; s > 0; s >>= 1) {
    if (t < s) red[t] = fmaxf(red[t], red[t + s]);
    __syncthreads();
  }
  m = red[0];
  __syncthreads();

  float sum = 0.f;
#pragma unroll
  for (int i = 0; i < 16; ++i) {
    v[i] = __expf(v[i] - m);
    sum += v[i];
  }
  red[t] = sum;
  __syncthreads();
  for (int s = 128; s > 0; s >>= 1) {
    if (t < s) red[t] += red[t + s];
    __syncthreads();
  }
  float inv = 1.0f / red[0];

  unsigned short* P = (unsigned short*)rp;  // bf16 over front half of the row
  v8us o0, o1;
#pragma unroll
  for (int i = 0; i < 8; ++i) o0[i] = f2bf(v[i] * inv);
#pragma unroll
  for (int i = 0; i < 8; ++i) o1[i] = f2bf(v[8 + i] * inv);
  ((v8us*)P)[2 * t + 0] = o0;
  ((v8us*)P)[2 * t + 1] = o1;
}

// ---------------------------------------------------------------------------
// Host side
// ---------------------------------------------------------------------------
extern "C" void kernel_launch(void* const* d_in, const int* in_sizes, int n_in,
                              void* d_out, int out_size, void* d_ws,
                              size_t ws_size, hipStream_t stream) {
  const float* x  = (const float*)d_in[0];
  const float* wq = (const float*)d_in[1];
  const float* wk = (const float*)d_in[2];
  const float* wv = (const float*)d_in[3];

  char* ws = (char*)d_ws;
  size_t off = 0;
  unsigned short* xh  = (unsigned short*)(ws + off); off += (size_t)SEQ * DIN * 2;   // 8 MB
  unsigned short* wqt = (unsigned short*)(ws + off); off += (size_t)DIN * DOUT * 2;  // 2 MB (W^T)
  unsigned short* wkt = (unsigned short*)(ws + off); off += (size_t)DIN * DOUT * 2;
  unsigned short* wvt = (unsigned short*)(ws + off); off += (size_t)DIN * DOUT * 2;
  unsigned short* qh  = (unsigned short*)(ws + off); off += (size_t)SEQ * DOUT * 2;  // 8 MB
  unsigned short* kh  = (unsigned short*)(ws + off); off += (size_t)SEQ * DOUT * 2;
  unsigned short* vh  = (unsigned short*)(ws + off); off += (size_t)SEQ * DOUT * 2;
  unsigned short* vt  = (unsigned short*)(ws + off); off += (size_t)SEQ * DOUT * 2;  // V^T
  float* scores = (float*)(ws + off);                off += (size_t)SEQ * SEQ * 4;   // 64 MB
  // P (bf16) lives in place over scores rows: element stride 2*SEQ ushorts/row

  dim3 blk(256);

  // x -> bf16 ; W -> bf16 transposed ([DOUT][DIN])
  cvt_f32_bf16<<<(SEQ * DIN / 8 + 255) / 256, blk, 0, stream>>>(x, xh, SEQ * DIN / 8);
  dim3 gW(DOUT / 64, DIN / 64);
  transpose_cvt_f32_bf16<<<gW, blk, 0, stream>>>(wq, wqt, DIN, DOUT);
  transpose_cvt_f32_bf16<<<gW, blk, 0, stream>>>(wk, wkt, DIN, DOUT);
  transpose_cvt_f32_bf16<<<gW, blk, 0, stream>>>(wv, wvt, DIN, DOUT);

  // Q, K, V = x @ W  == NT(x, W^T)  (bf16 out)
  dim3 gQKV(DOUT / 128, SEQ / 128);
  gemm_nt_bf16_wmma<true><<<gQKV, blk, 0, stream>>>(xh, DIN, wqt, DIN, qh, DOUT, DIN, 1.0f);
  gemm_nt_bf16_wmma<true><<<gQKV, blk, 0, stream>>>(xh, DIN, wkt, DIN, kh, DOUT, DIN, 1.0f);
  gemm_nt_bf16_wmma<true><<<gQKV, blk, 0, stream>>>(xh, DIN, wvt, DIN, vh, DOUT, DIN, 1.0f);

  // V^T for the P@V GEMM
  dim3 gV(DOUT / 64, SEQ / 64);
  transpose_bf16<<<gV, blk, 0, stream>>>(vh, vt, SEQ, DOUT);

  // scores = (Q @ K^T) * 1/sqrt(1024)  == NT(Q, K)  (f32 out)
  dim3 gS(SEQ / 128, SEQ / 128);
  gemm_nt_bf16_wmma<false><<<gS, blk, 0, stream>>>(qh, DOUT, kh, DOUT, scores, SEQ, DOUT, 0.03125f);

  // softmax rows, bf16 P written in place
  softmax_row_inplace<<<SEQ, blk, 0, stream>>>(scores, SEQ);

  // context = P @ V == NT(P, V^T)  (f32 out -> d_out)
  dim3 gC(DOUT / 128, SEQ / 128);
  gemm_nt_bf16_wmma<false><<<gC, blk, 0, stream>>>(
      (const unsigned short*)scores, 2 * SEQ, vt, SEQ, d_out, DOUT, SEQ, 1.0f);
}
